// AllGatherMoe_36816459661327
// MI455X (gfx1250) — matrix operand order repaired and checked by reading, not compiled, not run on hardware
//
#include <hip/hip_runtime.h>
#include <hip/hip_bf16.h>

typedef __attribute__((ext_vector_type(16))) __bf16 v16bf;
typedef __attribute__((ext_vector_type(8)))  __bf16 v8bf;
typedef __attribute__((ext_vector_type(2)))  __bf16 v2bf;
typedef __attribute__((ext_vector_type(8)))  float  v8f;

#define K_DIM   1024
#define N_OUT   1024      // output columns (= N/2)
#define NGEMM   2048      // gemm columns of weight
#define NEXP    8
#define NROWS   32768     // M * topk
#define TM      128       // rows per block tile
#define TN      64        // output cols per block tile (=> 128 gemm cols)
#define KC      32        // K step (bf16 WMMA K)
#define ASTR    40        // LDS stride in halfs (80B, 16B aligned, conflict-free frag reads)
#define BSTR    40
#define MAXTILES 264      // >= 32768/128 + 7

// meta (ints): [0..8] expert row offsets, [9..17] tile prefix sums, [18..25] scatter counters
#define META_OFF 0
#define META_TIL 9
#define META_CNT 18

__global__ void k_count(const int* __restrict__ ids, int* __restrict__ meta) {
    __shared__ int cnt[NEXP];
    const int t = threadIdx.x;
    if (t < NEXP) cnt[t] = 0;
    __syncthreads();
    for (int i = t; i < NROWS; i += blockDim.x) {
        atomicAdd(&cnt[ids[i]], 1);
    }
    __syncthreads();
    if (t == 0) {
        int off = 0, tc = 0;
        for (int e = 0; e < NEXP; e++) {
            meta[META_OFF + e] = off;
            meta[META_CNT + e] = off;          // scatter counters, reset every launch
            meta[META_TIL + e] = tc;
            tc  += (cnt[e] + TM - 1) / TM;
            off += cnt[e];
        }
        meta[META_OFF + NEXP] = off;           // = NROWS
        meta[META_TIL + NEXP] = tc;            // total tiles
    }
}

__global__ void k_scatter(const int* __restrict__ ids, int* __restrict__ meta,
                          int* __restrict__ sorted) {
    const int i = blockIdx.x * blockDim.x + threadIdx.x;
    if (i < NROWS) {
        const int e = ids[i];
        const int p = atomicAdd(&meta[META_CNT + e], 1);
        sorted[p] = i;
    }
}

__global__ __launch_bounds__(256)
void k_moe(const float* __restrict__ hidden,     // [16384, 1024] f32
           const float* __restrict__ w,          // [8, 1024, 2048] f32
           const int*   __restrict__ meta,
           const int*   __restrict__ sorted,
           float*       __restrict__ out) {      // [32768, 1024] f32
    __shared__ __align__(16) __bf16 As[TM * ASTR];        // 128 rows x 32 k (padded)
    __shared__ __align__(16) __bf16 Bs[2 * TN * BSTR];    // 128 n x 32 k, transposed
    __shared__ int sRows[TM];

    const int tid = threadIdx.x;
    const int bx  = blockIdx.x;
    const int gn0 = blockIdx.y * TN;

    if (bx >= meta[META_TIL + NEXP]) return;

    // find expert: largest e with tilePrefix[e] <= bx
    int e = 0;
#pragma unroll
    for (int i = 1; i < NEXP; i++)
        if (bx >= meta[META_TIL + i]) e = i;

    const int segStart = meta[META_OFF + e];
    const int segEnd   = meta[META_OFF + e + 1];
    const int rowStart = segStart + (bx - meta[META_TIL + e]) * TM;

    if (tid < TM) {
        const int gi = rowStart + tid;
        sRows[tid] = (gi < segEnd) ? sorted[gi] : sorted[segStart];  // pad w/ valid row
    }
    __syncthreads();                        // publish sRows

    const float* wptr = w + (size_t)e * K_DIM * NGEMM;

    const int lane = tid & 31;
    const int wave = tid >> 5;
    const int wr   = wave & 3;          // 0..3 : row quadrant (32 rows each)
    const int wc   = wave >> 2;         // 0..1 : col half
    const int l16  = lane >> 4;         // 0/1  : lane group
    const int ln   = lane & 15;

    v8f acc[2][4];                      // [row subtile][0,1 gate | 2,3 up]
#pragma unroll
    for (int i = 0; i < 2; i++)
#pragma unroll
        for (int j = 0; j < 4; j++)
#pragma unroll
            for (int r = 0; r < 8; r++) acc[i][j][r] = 0.0f;

    // ---- hoisted A-gather base pointer (loop-invariant) ----
    const int arow  = tid >> 1;         // A-staging row 0..127
    const int ahalf = tid & 1;          // A-staging k-half (16 halfs each)
    const float* aSrc = hidden + (size_t)(sRows[arow] >> 1) * K_DIM + ahalf * 16;

    // ---- B-staging indices: thread owns contiguous 16-wide K range of one column ----
    const int nn    = tid & 127;        // gemm-col within tile
    const int khalf = (tid >> 7) * 16;  // K base: 0 or 16
    const int gcol  = (nn < TN) ? (gn0 + nn) : (N_OUT + gn0 + (nn - TN));
    const float* bSrc = wptr + (size_t)khalf * NGEMM + gcol;

    for (int kk = 0; kk < K_DIM; kk += KC) {
        __syncthreads();   // LDS reuse guard

        // ---- stage A: gather rows, convert f32 -> bf16 (v_cvt_pk_bf16_f32) ----
        {
            const float* src = aSrc + kk;
            const float4 f0 = ((const float4*)src)[0];
            const float4 f1 = ((const float4*)src)[1];
            const float4 f2 = ((const float4*)src)[2];
            const float4 f3 = ((const float4*)src)[3];
            v8bf p0, p1;
            p0[0]=(__bf16)f0.x; p0[1]=(__bf16)f0.y; p0[2]=(__bf16)f0.z; p0[3]=(__bf16)f0.w;
            p0[4]=(__bf16)f1.x; p0[5]=(__bf16)f1.y; p0[6]=(__bf16)f1.z; p0[7]=(__bf16)f1.w;
            p1[0]=(__bf16)f2.x; p1[1]=(__bf16)f2.y; p1[2]=(__bf16)f2.z; p1[3]=(__bf16)f2.w;
            p1[4]=(__bf16)f3.x; p1[5]=(__bf16)f3.y; p1[6]=(__bf16)f3.z; p1[7]=(__bf16)f3.w;
            *(v8bf*)&As[arow * ASTR + ahalf * 16]     = p0;
            *(v8bf*)&As[arow * ASTR + ahalf * 16 + 8] = p1;
        }

        // ---- stage B: coalesced f32 reads; paired cvt_pk + b32 transposed stores ----
#pragma unroll
        for (int i = 0; i < 8; i++) {
            const float f0 = bSrc[(size_t)(kk + 2 * i)     * NGEMM];
            const float f1 = bSrc[(size_t)(kk + 2 * i + 1) * NGEMM];
            v2bf p;
            p[0] = (__bf16)f0;
            p[1] = (__bf16)f1;
            *(v2bf*)&Bs[nn * BSTR + khalf + 2 * i] = p;   // 4B-aligned (BSTR even)
        }
        // prefetch next K-step weight rows into cache (global_prefetch_b8)
        if (kk + KC < K_DIM) {
            __builtin_prefetch(&bSrc[(size_t)(kk + KC) * NGEMM], 0, 1);
        }

        __syncthreads();

        // ---- load fragments ----
        v16bf af[2];
#pragma unroll
        for (int rs = 0; rs < 2; rs++) {
            const int row = wr * 32 + rs * 16 + ln;
            const int kb  = l16 * 8;                       // K base 0 or 8
            const v8bf lo = *(const v8bf*)&As[row * ASTR + kb];        // K kb..kb+7
            const v8bf hi = *(const v8bf*)&As[row * ASTR + kb + 16];   // K kb+16..kb+23
#pragma unroll
            for (int j = 0; j < 8; j++) { af[rs][j] = lo[j]; af[rs][j + 8] = hi[j]; }
        }
        v16bf bfm[4];
#pragma unroll
        for (int cs = 0; cs < 4; cs++) {
            const int nsub = (cs < 2) ? (wc * 2 + cs) : (4 + wc * 2 + (cs - 2));
            const int n  = nsub * 16 + ln;
            const int kb = l16 * 8;                        // K base 0 or 8
            const v8bf lo = *(const v8bf*)&Bs[n * BSTR + kb];
            const v8bf hi = *(const v8bf*)&Bs[n * BSTR + kb + 16];
#pragma unroll
            for (int j = 0; j < 8; j++) { bfm[cs][j] = lo[j]; bfm[cs][j + 8] = hi[j]; }
        }

        // ---- 8 WMMAs per wave ----
#pragma unroll
        for (int rs = 0; rs < 2; rs++)
#pragma unroll
            for (int cs = 0; cs < 4; cs++)
                acc[rs][cs] = __builtin_amdgcn_wmma_f32_16x16x32_bf16(
                    false, af[rs], false, bfm[cs], (short)0, acc[rs][cs], false, false);
    }

    // ---- fused SwiGLU epilogue: gate = acc[.][0,1], up = acc[.][2,3] ----
#pragma unroll
    for (int rs = 0; rs < 2; rs++) {
        // hoist row ids + validity (independent of cs)
        int  rid[8];
        bool ok[8];
#pragma unroll
        for (int r = 0; r < 8; r++) {
            const int ml = wr * 32 + rs * 16 + r + l16 * 8;
            ok[r]  = (rowStart + ml) < segEnd;
            rid[r] = sRows[ml];
        }
#pragma unroll
        for (int cs = 0; cs < 2; cs++) {
            const v8f g = acc[rs][cs];
            const v8f u = acc[rs][cs + 2];
            const int n = gn0 + (wc * 2 + cs) * 16 + ln;
#pragma unroll
            for (int r = 0; r < 8; r++) {
                if (ok[r]) {
                    const float gv = g[r];
                    // silu via hardware reciprocal (v_rcp_f32), avoids IEEE divide expansion
                    const float s = gv * __builtin_amdgcn_rcpf(1.0f + __expf(-gv));
                    out[(size_t)rid[r] * N_OUT + n] = s * u[r];
                }
            }
        }
    }
}

extern "C" void kernel_launch(void* const* d_in, const int* in_sizes, int n_in,
                              void* d_out, int out_size, void* d_ws, size_t ws_size,
                              hipStream_t stream) {
    const float* hidden = (const float*)d_in[0];   // [16384,1024] f32
    const float* w      = (const float*)d_in[1];   // [8,1024,2048] f32
    const int*   ids    = (const int*)d_in[2];     // [16384,2] flat 32768
    float*       out    = (float*)d_out;           // [32768,1024] f32

    int* meta   = (int*)d_ws;                      // 32 ints
    int* sorted = (int*)((char*)d_ws + 256);       // 32768 ints

    k_count  <<<1, 256, 0, stream>>>(ids, meta);
    k_scatter<<<NROWS / 256, 256, 0, stream>>>(ids, meta, sorted);
    k_moe    <<<dim3(MAXTILES, N_OUT / TN), 256, 0, stream>>>(hidden, w, meta, sorted, out);
}
